// Block_54872502174070
// MI455X (gfx1250) — matrix operand order repaired and checked by reading, not compiled, not run on hardware
//
#include <hip/hip_runtime.h>
#include <math.h>

// ---------------------------------------------------------------------------
// Types for CDNA5 WMMA (wave32)
// ---------------------------------------------------------------------------
typedef __bf16 bf16_t;
typedef __attribute__((ext_vector_type(16))) __bf16 v16bf;
typedef __attribute__((ext_vector_type(8)))  __bf16 v8bf;
typedef __attribute__((ext_vector_type(8)))  float  v8f;

#define DEVINL __device__ __forceinline__

DEVINL void bsplit(float v, bf16_t* hi, bf16_t* lo) {
  bf16_t h = (bf16_t)v;
  *hi = h;
  *lo = (bf16_t)(v - (float)h);
}

// ---------------------------------------------------------------------------
// WMMA fragment loaders (per ISA 7.12.2 layouts, wave32)
// A 16x32 bf16: lane<16 -> row M=lane, elems = K{k0+0..7, k0+16..23}
//               lane>=16 -> row M=lane-16, elems = K{k0+8..15, k0+24..31}
// B 32x16 bf16 (loaded from B^T, N x K row-major):
//               lane<16 -> col N=lane, elems = K{k0+0..7, k0+16..23}
//               lane>=16 -> col N=lane-16, elems = K{k0+8..15, k0+24..31}
// ---------------------------------------------------------------------------
DEVINL v16bf load_a_frag(const bf16_t* __restrict__ A, int m0, int Kp, int k0, int lane) {
  int half = lane >> 4;
  int row  = m0 + (lane & 15);
  const bf16_t* p = A + (size_t)row * Kp + k0 + half * 8;
  v8bf a0 = *(const v8bf*)(p);
  v8bf a1 = *(const v8bf*)(p + 16);
  v16bf r;
#pragma unroll
  for (int i = 0; i < 8; ++i) { r[i] = a0[i]; r[i + 8] = a1[i]; }
  return r;
}

DEVINL v16bf load_b_frag(const bf16_t* __restrict__ BT, int n0, int Kp, int k0, int lane) {
  int half = lane >> 4;
  int row  = n0 + (lane & 15);
  const bf16_t* p = BT + (size_t)row * Kp + k0 + half * 8;
  v8bf b0 = *(const v8bf*)(p);
  v8bf b1 = *(const v8bf*)(p + 16);
  v16bf r;
#pragma unroll
  for (int i = 0; i < 8; ++i) { r[i] = b0[i]; r[i + 8] = b1[i]; }
  return r;
}

// ---------------------------------------------------------------------------
// Generic WMMA GEMM: C = A(M x Kp) * B(Kp x N), bf16 hi/lo split, f32 accum.
// 3 WMMAs per k-step: Ah*Bh + Ah*Bl + Al*Bh.
// Block = 256 threads = 8 waves; each wave computes one 16x16 tile;
// grid = (N/16, M/128). Waves in a block share the N-tile (B reuse in L0/L2).
// ---------------------------------------------------------------------------
enum { EPI_F32 = 0, EPI_PROJ = 1, EPI_GELU = 2, EPI_RES = 3, EPI_BFT = 4 };

template <int EPI>
__global__ __launch_bounds__(256)
void wmma_gemm(const bf16_t* __restrict__ Ahi, const bf16_t* __restrict__ Alo,
               const bf16_t* __restrict__ BThi, const bf16_t* __restrict__ BTlo,
               int M, int N, int Kp, int ldo,
               const float* __restrict__ bias, const float* __restrict__ res,
               float* __restrict__ out, float* __restrict__ out2,
               bf16_t* __restrict__ ohi, bf16_t* __restrict__ olo)
{
  int wave = threadIdx.x >> 5;
  int lane = threadIdx.x & 31;
  int n0 = blockIdx.x * 16;
  int m0 = (blockIdx.y * 8 + wave) * 16;
  if (m0 >= M || n0 >= N) return;

  v8f acc = {0.f, 0.f, 0.f, 0.f, 0.f, 0.f, 0.f, 0.f};
  for (int k0 = 0; k0 < Kp; k0 += 32) {
    v16bf ah = load_a_frag(Ahi, m0, Kp, k0, lane);
    v16bf al = load_a_frag(Alo, m0, Kp, k0, lane);
    v16bf bh = load_b_frag(BThi, n0, Kp, k0, lane);
    v16bf bl = load_b_frag(BTlo, n0, Kp, k0, lane);
    acc = __builtin_amdgcn_wmma_f32_16x16x32_bf16(false, ah, false, bh, (short)0, acc, false, false);
    acc = __builtin_amdgcn_wmma_f32_16x16x32_bf16(false, ah, false, bl, (short)0, acc, false, false);
    acc = __builtin_amdgcn_wmma_f32_16x16x32_bf16(false, al, false, bh, (short)0, acc, false, false);
  }

  // C/D layout: lane<16 -> N=lane, VGPR j -> M=j; lane>=16 -> N=lane-16, M=8+j
  int half = lane >> 4;
  int nn = n0 + (lane & 15);
#pragma unroll
  for (int j = 0; j < 8; ++j) {
    int mm = m0 + half * 8 + j;
    float v = acc[j];
    size_t oi = (size_t)mm * ldo + nn;
    if constexpr (EPI == EPI_F32) {
      if (bias) v += bias[nn];
      out[oi] = v;
    } else if constexpr (EPI == EPI_PROJ) {
      v += bias[nn];
      out[oi]  = v;            // o (biformer input)
      out2[oi] = v + res[oi];  // xr = o + x
    } else if constexpr (EPI == EPI_GELU) {
      v += bias[nn];
      float g = 0.5f * v * (1.0f + erff(v * 0.70710678118654752f));
      bsplit(g, &ohi[oi], &olo[oi]);
    } else if constexpr (EPI == EPI_RES) {
      v += bias[nn] + res[oi];
      out[oi] = v;
    } else if constexpr (EPI == EPI_BFT) {
      v += bias[nn];
      int bb = mm / 784;
      int pp = mm - bb * 784;
      out[(size_t)bb * 50176 + (size_t)nn * 784 + pp] = v; // (B, 64, 28, 28)
    }
  }
}

// ---------------------------------------------------------------------------
// Weight prep: W(K x N) f32 -> B^T (N x Kp) bf16 hi/lo (zero-padded K..Kp)
// transpose=0 means W already stored as (N x K).
// ---------------------------------------------------------------------------
__global__ __launch_bounds__(256)
void split_wT_kernel(const float* __restrict__ W, bf16_t* __restrict__ hi,
                     bf16_t* __restrict__ lo, int K, int N, int Kp, int transpose)
{
  long long i = (long long)blockIdx.x * 256 + threadIdx.x;
  if (i >= (long long)N * Kp) return;
  int k = (int)(i % Kp);
  int n = (int)(i / Kp);
  float v = 0.f;
  if (k < K) v = transpose ? W[(size_t)k * N + n] : W[(size_t)n * K + k];
  bsplit(v, &hi[i], &lo[i]);
}

// ---------------------------------------------------------------------------
// LayerNorm over last dim C=784, output bf16 hi/lo with row stride Cp=800
// (pad columns zeroed). One block (256 thr) per row.
// ---------------------------------------------------------------------------
__global__ __launch_bounds__(256)
void ln_split_kernel(const float* __restrict__ X, const float* __restrict__ g,
                     const float* __restrict__ bt, bf16_t* __restrict__ hi,
                     bf16_t* __restrict__ lo, int C, int Cp)
{
  int row = blockIdx.x;
  const float* x = X + (size_t)row * C;
  __shared__ float red[256];
  float s = 0.f;
  for (int c = threadIdx.x; c < C; c += 256) s += x[c];
  red[threadIdx.x] = s; __syncthreads();
  for (int o = 128; o > 0; o >>= 1) {
    if (threadIdx.x < o) red[threadIdx.x] += red[threadIdx.x + o];
    __syncthreads();
  }
  float mean = red[0] / C;
  __syncthreads();
  float vs = 0.f;
  for (int c = threadIdx.x; c < C; c += 256) { float d = x[c] - mean; vs += d * d; }
  red[threadIdx.x] = vs; __syncthreads();
  for (int o = 128; o > 0; o >>= 1) {
    if (threadIdx.x < o) red[threadIdx.x] += red[threadIdx.x + o];
    __syncthreads();
  }
  float rstd = rsqrtf(red[0] / C + 1e-5f);
  bf16_t* ph = hi + (size_t)row * Cp;
  bf16_t* pl = lo + (size_t)row * Cp;
  for (int c = threadIdx.x; c < Cp; c += 256) {
    float v = (c < C) ? ((x[c] - mean) * rstd * g[c] + bt[c]) : 0.f;
    bsplit(v, &ph[c], &pl[c]);
  }
}

// ---------------------------------------------------------------------------
// Standard MHA (B=128, NH=8, N=64 tokens, d=98). One block per (b,h),
// 64 threads (one per query token). K,V staged in LDS (~49 KB of 320 KB WGP).
// Writes attention output directly as bf16 hi/lo rows (stride 800) for proj.
// ---------------------------------------------------------------------------
__global__ __launch_bounds__(64)
void mha_kernel(const float* __restrict__ qkv, bf16_t* __restrict__ ahi,
                bf16_t* __restrict__ alo)
{
  int b = blockIdx.x >> 3;
  int h = blockIdx.x & 7;
  __shared__ float sk[64][98];
  __shared__ float sv[64][98];
  const float* base = qkv + (size_t)b * 64 * 2352;
  for (int i = threadIdx.x; i < 64 * 98; i += 64) {
    int m = i / 98, d = i - m * 98;
    sk[m][d] = base[(size_t)m * 2352 + 784  + h * 98 + d];
    sv[m][d] = base[(size_t)m * 2352 + 1568 + h * 98 + d];
  }
  __syncthreads();

  int n = threadIdx.x;
  const float* qp = base + (size_t)n * 2352 + h * 98;
  float s[64];
#pragma unroll
  for (int m = 0; m < 64; ++m) s[m] = 0.f;
  for (int d = 0; d < 98; ++d) {
    float qd = qp[d];
#pragma unroll
    for (int m = 0; m < 64; ++m) s[m] += qd * sk[m][d];
  }
  const float scale = 0.1010152544552211f; // 98^-0.5
  float mx = -3.4e38f;
#pragma unroll
  for (int m = 0; m < 64; ++m) { s[m] *= scale; mx = fmaxf(mx, s[m]); }
  float sum = 0.f;
#pragma unroll
  for (int m = 0; m < 64; ++m) { s[m] = expf(s[m] - mx); sum += s[m]; }
  float inv = 1.f / sum;
#pragma unroll
  for (int m = 0; m < 64; ++m) s[m] *= inv;

  size_t row = (size_t)b * 64 + n;
  bf16_t* ph = ahi + row * 800 + h * 98;
  bf16_t* pl = alo + row * 800 + h * 98;
  for (int d = 0; d < 98; ++d) {
    float acc = 0.f;
#pragma unroll
    for (int m = 0; m < 64; ++m) acc += s[m] * sv[m][d];
    bsplit(acc, &ph[d], &pl[d]);
  }
  if (h == 0) { // zero K-padding columns 784..799 of this row
    for (int c = 784; c < 800; ++c) {
      ahi[row * 800 + c] = (bf16_t)0.f;
      alo[row * 800 + c] = (bf16_t)0.f;
    }
  }
}

// ---------------------------------------------------------------------------
// BiFormer helpers
// ---------------------------------------------------------------------------
// o (B,64,784) -> pixel-major A (B*784, 64) bf16 hi/lo
__global__ __launch_bounds__(256)
void pix_split_kernel(const float* __restrict__ o, bf16_t* __restrict__ hi,
                      bf16_t* __restrict__ lo)
{
  size_t i = (size_t)blockIdx.x * 256 + threadIdx.x; // over 128*784*64
  int c = (int)(i & 63);
  size_t rest = i >> 6;
  int p = (int)(rest % 784);
  int b = (int)(rest / 784);
  float v = o[((size_t)b * 64 + c) * 784 + p];
  bsplit(v, &hi[i], &lo[i]);
}

// Region mean pool: q_r/k_r (B, 64, 49)
__global__ __launch_bounds__(64)
void bf_pool_kernel(const float* __restrict__ qkvbf, float* __restrict__ q_r,
                    float* __restrict__ k_r)
{
  int r = blockIdx.x;   // 0..48
  int b = blockIdx.y;   // 0..127
  int c = threadIdx.x;  // 0..63
  int rh = r / 7, rw = r - rh * 7;
  float sq = 0.f, sk = 0.f;
  for (int i = 0; i < 4; ++i)
    for (int j = 0; j < 4; ++j) {
      int pix = (rh * 4 + i) * 28 + rw * 4 + j;
      const float* pp = qkvbf + ((size_t)b * 784 + pix) * 192;
      sq += pp[c];
      sk += pp[64 + c];
    }
  q_r[(size_t)b * 3136 + c * 49 + r] = sq * 0.0625f;
  k_r[(size_t)b * 3136 + c * 49 + r] = sk * 0.0625f;
}

// Per-batch routing scores a_r = q_r^T k_r (49x49) + top-4 per row
__global__ __launch_bounds__(64)
void bf_topk_kernel(const float* __restrict__ q_r, const float* __restrict__ k_r,
                    int* __restrict__ idx)
{
  int b = blockIdx.x;
  __shared__ float sq[64 * 49];
  __shared__ float sk2[64 * 49];
  for (int i = threadIdx.x; i < 64 * 49; i += 64) {
    sq[i]  = q_r[(size_t)b * 3136 + i];
    sk2[i] = k_r[(size_t)b * 3136 + i];
  }
  __syncthreads();
  int r = threadIdx.x;
  if (r >= 49) return;
  float a[49];
  for (int s = 0; s < 49; ++s) {
    float acc = 0.f;
#pragma unroll
    for (int c = 0; c < 64; ++c) acc += sq[c * 49 + r] * sk2[c * 49 + s];
    a[s] = acc;
  }
  int chosen[4];
  for (int t = 0; t < 4; ++t) {
    float best = -3.4e38f;
    int bi = 0;
    for (int s = 0; s < 49; ++s) {
      bool skip = false;
      for (int u = 0; u < t; ++u) if (chosen[u] == s) skip = true;
      if (!skip && a[s] > best) { best = a[s]; bi = s; }
    }
    chosen[t] = bi;
    idx[((size_t)b * 49 + r) * 4 + t] = bi;
  }
}

// Routed attention: one thread per (b,h,r,p); 4 regions x 16 pos gathered KV.
__global__ __launch_bounds__(64)
void bf_attn_kernel(const float* __restrict__ qkvbf, const int* __restrict__ idx,
                    float* __restrict__ bfg)
{
  int b = blockIdx.z;
  int h = blockIdx.y;
  int r = blockIdx.x * 4 + (threadIdx.x >> 4);
  int p = threadIdx.x & 15;
  if (r >= 49) return;
  int rh = r / 7, rw = r - rh * 7;
  int pix = (rh * 4 + (p >> 2)) * 28 + rw * 4 + (p & 3);

  const float* qp = qkvbf + ((size_t)b * 784 + pix) * 192 + h * 8;
  float q[8];
#pragma unroll
  for (int d = 0; d < 8; ++d) q[d] = qp[d] * 0.125f; // scale = 64^-0.5

  float s[64];
  int spx[64];
  for (int t = 0; t < 4; ++t) {
    int sr = idx[((size_t)b * 49 + r) * 4 + t];
    int sh = sr / 7, sw = sr - sh * 7;
    for (int sp = 0; sp < 16; ++sp) {
      int spix = (sh * 4 + (sp >> 2)) * 28 + sw * 4 + (sp & 3);
      spx[t * 16 + sp] = spix;
      const float* kp = qkvbf + ((size_t)b * 784 + spix) * 192 + 64 + h * 8;
      float a = 0.f;
#pragma unroll
      for (int d = 0; d < 8; ++d) a += q[d] * kp[d];
      s[t * 16 + sp] = a;
    }
  }
  float mx = -3.4e38f;
#pragma unroll
  for (int m = 0; m < 64; ++m) mx = fmaxf(mx, s[m]);
  float sum = 0.f;
#pragma unroll
  for (int m = 0; m < 64; ++m) { s[m] = expf(s[m] - mx); sum += s[m]; }
  float inv = 1.f / sum;

  float o[8];
#pragma unroll
  for (int d = 0; d < 8; ++d) o[d] = 0.f;
  for (int m = 0; m < 64; ++m) {
    float pm = s[m] * inv;
    const float* vp = qkvbf + ((size_t)b * 784 + spx[m]) * 192 + 128 + h * 8;
#pragma unroll
    for (int d = 0; d < 8; ++d) o[d] += pm * vp[d];
  }
#pragma unroll
  for (int d = 0; d < 8; ++d)
    bfg[(size_t)b * 50176 + (size_t)(h * 8 + d) * 784 + pix] = o[d];
}

// LePE (3x3 depthwise on v) + attention grid -> pixel-major bf16 hi/lo
__global__ __launch_bounds__(256)
void bf_lepe_kernel(const float* __restrict__ qkvbf, const float* __restrict__ w,
                    const float* __restrict__ bias, const float* __restrict__ bfg,
                    bf16_t* __restrict__ a2hi, bf16_t* __restrict__ a2lo)
{
  size_t i = (size_t)blockIdx.x * 256 + threadIdx.x; // over 128*784*64
  int c = (int)(i & 63);
  size_t rest = i >> 6;
  int p = (int)(rest % 784);
  int b = (int)(rest / 784);
  int y = p / 28, x = p - y * 28;
  float acc = bias[c];
#pragma unroll
  for (int dy = -1; dy <= 1; ++dy)
#pragma unroll
    for (int dx = -1; dx <= 1; ++dx) {
      int yy = y + dy, xx = x + dx;
      if (yy < 0 || yy >= 28 || xx < 0 || xx >= 28) continue;
      float vv = qkvbf[((size_t)b * 784 + yy * 28 + xx) * 192 + 128 + c];
      acc += vv * w[c * 9 + (dy + 1) * 3 + (dx + 1)];
    }
  acc += bfg[(size_t)b * 50176 + (size_t)c * 784 + p];
  bsplit(acc, &a2hi[i], &a2lo[i]);
}

// ---------------------------------------------------------------------------
// Host orchestration
// ---------------------------------------------------------------------------
extern "C" void kernel_launch(void* const* d_in, const int* in_sizes, int n_in,
                              void* d_out, int out_size, void* d_ws, size_t ws_size,
                              hipStream_t stream) {
  (void)in_sizes; (void)n_in; (void)out_size; (void)ws_size;

  const float* x      = (const float*)d_in[0];
  const float* n1g    = (const float*)d_in[1];
  const float* n1b    = (const float*)d_in[2];
  const float* qkv_w  = (const float*)d_in[3];
  const float* proj_w = (const float*)d_in[4];
  const float* proj_b = (const float*)d_in[5];
  const float* n2g    = (const float*)d_in[6];
  const float* n2b    = (const float*)d_in[7];
  const float* fc1_w  = (const float*)d_in[8];
  const float* fc1_b  = (const float*)d_in[9];
  const float* fc2_w  = (const float*)d_in[10];
  const float* fc2_b  = (const float*)d_in[11];
  const float* bqkv_w = (const float*)d_in[12];
  const float* bqkv_b = (const float*)d_in[13];
  const float* lepe_w = (const float*)d_in[14];
  const float* lepe_b = (const float*)d_in[15];
  const float* bout_w = (const float*)d_in[16];
  const float* bout_b = (const float*)d_in[17];

  float* out_main = (float*)d_out;            // (128, 64, 784)
  float* out_bf   = out_main + 6422528;       // (128, 64, 28, 28)

  char* ws = (char*)d_ws;
  // Phase-aliased workspace layout (all region sizes multiples of 256 B):
  const size_t SZ_HHALF = (size_t)8192 * 800 * 2;     // 13,107,200 (one bf16 plane)
  const size_t R1 = 0;                                // h / attn / h2 (hi+lo)
  const size_t R2 = R1 + 2 * SZ_HHALF;                // qkv f32 / qkvbf / g head
  const size_t R3 = R2 + (size_t)8192 * 2352 * 4;     // o / bfg / g tail
  const size_t R4 = R3 + (size_t)8192 * 784 * 4;      // Apix / A2 (hi+lo)
  const size_t XRo = R4 + (size_t)100352 * 64 * 2 * 2;
  size_t off = XRo + (size_t)8192 * 784 * 4;
  const size_t WQ  = off; off += 2 * (size_t)2352 * 800 * 2;
  const size_t WP  = off; off += 2 * (size_t)784 * 800 * 2;
  const size_t W1  = off; off += 2 * (size_t)3136 * 800 * 2;
  const size_t W2  = off; off += 2 * (size_t)784 * 3136 * 2;
  const size_t WBQ = off; off += 2 * (size_t)192 * 64 * 2;
  const size_t WBO = off; off += 2 * (size_t)64 * 64 * 2;
  const size_t QRo = off; off += (size_t)128 * 64 * 49 * 4;
  const size_t KRo = off; off += (size_t)128 * 64 * 49 * 4;
  const size_t IXo = off; off += (size_t)128 * 49 * 4 * 4;

  bf16_t* h_hi   = (bf16_t*)(ws + R1);
  bf16_t* h_lo   = (bf16_t*)(ws + R1 + SZ_HHALF);
  bf16_t* at_hi  = h_hi;                    // aliases (disjoint lifetimes)
  bf16_t* at_lo  = h_lo;
  bf16_t* h2_hi  = h_hi;
  bf16_t* h2_lo  = h_lo;
  float*  qkv_f  = (float*)(ws + R2);
  float*  qkvbf  = (float*)(ws + R2);
  bf16_t* g_hi   = (bf16_t*)(ws + R2);
  bf16_t* g_lo   = (bf16_t*)(ws + R2 + (size_t)8192 * 3136 * 2);
  float*  o_f    = (float*)(ws + R3);
  float*  bfg    = (float*)(ws + R3);
  bf16_t* ap_hi  = (bf16_t*)(ws + R4);
  bf16_t* ap_lo  = (bf16_t*)(ws + R4 + (size_t)100352 * 64 * 2);
  bf16_t* a2_hi  = ap_hi;
  bf16_t* a2_lo  = ap_lo;
  float*  xr     = (float*)(ws + XRo);
  bf16_t* wq_hi  = (bf16_t*)(ws + WQ);
  bf16_t* wq_lo  = wq_hi + (size_t)2352 * 800;
  bf16_t* wp_hi  = (bf16_t*)(ws + WP);
  bf16_t* wp_lo  = wp_hi + (size_t)784 * 800;
  bf16_t* w1_hi  = (bf16_t*)(ws + W1);
  bf16_t* w1_lo  = w1_hi + (size_t)3136 * 800;
  bf16_t* w2_hi  = (bf16_t*)(ws + W2);
  bf16_t* w2_lo  = w2_hi + (size_t)784 * 3136;
  bf16_t* wbq_hi = (bf16_t*)(ws + WBQ);
  bf16_t* wbq_lo = wbq_hi + (size_t)192 * 64;
  bf16_t* wbo_hi = (bf16_t*)(ws + WBO);
  bf16_t* wbo_lo = wbo_hi + (size_t)64 * 64;
  float*  q_r    = (float*)(ws + QRo);
  float*  k_r    = (float*)(ws + KRo);
  int*    idx    = (int*)(ws + IXo);

  // --- Weight prep: B^T bf16 hi/lo (K padded to 32-multiples) ---
  split_wT_kernel<<<(2352 * 800 + 255) / 256, 256, 0, stream>>>(qkv_w,  wq_hi,  wq_lo,  784, 2352, 800, 1);
  split_wT_kernel<<<(784 * 800 + 255) / 256, 256, 0, stream>>>(proj_w, wp_hi,  wp_lo,  784, 784,  800, 1);
  split_wT_kernel<<<(3136 * 800 + 255) / 256, 256, 0, stream>>>(fc1_w,  w1_hi,  w1_lo,  784, 3136, 800, 1);
  split_wT_kernel<<<(784 * 3136 + 255) / 256, 256, 0, stream>>>(fc2_w,  w2_hi,  w2_lo,  3136, 784, 3136, 1);
  split_wT_kernel<<<(192 * 64 + 255) / 256, 256, 0, stream>>>(bqkv_w, wbq_hi, wbq_lo, 64, 192, 64, 0);
  split_wT_kernel<<<(64 * 64 + 255) / 256, 256, 0, stream>>>(bout_w, wbo_hi, wbo_lo, 64, 64, 64, 0);

  // --- LN1 -> h (bf16 hi/lo, rows 8192, stride 800) ---
  ln_split_kernel<<<8192, 256, 0, stream>>>(x, n1g, n1b, h_hi, h_lo, 784, 800);

  // --- qkv GEMM: (8192 x 800) * (800 x 2352), no bias ---
  wmma_gemm<EPI_F32><<<dim3(147, 64), 256, 0, stream>>>(
      h_hi, h_lo, wq_hi, wq_lo, 8192, 2352, 800, 2352,
      nullptr, nullptr, qkv_f, nullptr, nullptr, nullptr);

  // --- MHA -> attention output bf16 hi/lo (stride 800) ---
  mha_kernel<<<1024, 64, 0, stream>>>(qkv_f, at_hi, at_lo);

  // --- proj GEMM: o = attn*Wp + b ; xr = o + x ---
  wmma_gemm<EPI_PROJ><<<dim3(49, 64), 256, 0, stream>>>(
      at_hi, at_lo, wp_hi, wp_lo, 8192, 784, 800, 784,
      proj_b, x, o_f, xr, nullptr, nullptr);

  // --- BiFormer branch ---
  pix_split_kernel<<<25088, 256, 0, stream>>>(o_f, ap_hi, ap_lo);
  wmma_gemm<EPI_F32><<<dim3(12, 784), 256, 0, stream>>>(
      ap_hi, ap_lo, wbq_hi, wbq_lo, 100352, 192, 64, 192,
      bqkv_b, nullptr, qkvbf, nullptr, nullptr, nullptr);
  bf_pool_kernel<<<dim3(49, 128), 64, 0, stream>>>(qkvbf, q_r, k_r);
  bf_topk_kernel<<<128, 64, 0, stream>>>(q_r, k_r, idx);
  bf_attn_kernel<<<dim3(13, 8, 128), 64, 0, stream>>>(qkvbf, idx, bfg);
  bf_lepe_kernel<<<25088, 256, 0, stream>>>(qkvbf, lepe_w, lepe_b, bfg, a2_hi, a2_lo);
  wmma_gemm<EPI_BFT><<<dim3(4, 784), 256, 0, stream>>>(
      a2_hi, a2_lo, wbo_hi, wbo_lo, 100352, 64, 64, 64,
      bout_b, nullptr, out_bf, nullptr, nullptr, nullptr);

  // --- MLP ---
  ln_split_kernel<<<8192, 256, 0, stream>>>(xr, n2g, n2b, h2_hi, h2_lo, 784, 800);
  wmma_gemm<EPI_GELU><<<dim3(196, 64), 256, 0, stream>>>(
      h2_hi, h2_lo, w1_hi, w1_lo, 8192, 3136, 800, 3136,
      fc1_b, nullptr, nullptr, nullptr, g_hi, g_lo);
  wmma_gemm<EPI_RES><<<dim3(49, 64), 256, 0, stream>>>(
      g_hi, g_lo, w2_hi, w2_lo, 8192, 784, 3136, 784,
      fc2_b, xr, out_main, nullptr, nullptr, nullptr);
}
